// TopKDecoder_61521111548291
// MI455X (gfx1250) — matrix operand order repaired and checked by reading, not compiled, not run on hardware
//
#include <hip/hip_runtime.h>
#include <hip/hip_bf16.h>

// ---------------------------------------------------------------------------
// CDNA5 (gfx1250) beam-search decoder.
// All GEMMs on v_wmma_f32_16x16x32_bf16; branchless inner loops with uniform
// base + 32-bit offsets (SADDR addressing, no 64-bit per-lane pointer adds,
// no spills); 2 N-tiles per wave for A reuse. Weights/encoder converted once
// to bf16 (W_out^T = 40MB, encT = 16MB -> L2-resident on 192MB L2).
// ---------------------------------------------------------------------------

typedef __attribute__((ext_vector_type(16))) __bf16      v16bf;
typedef __attribute__((ext_vector_type(8)))  float       v8f;
typedef __attribute__((ext_vector_type(4)))  unsigned int v4u;

union V16 { v16bf b; v4u q[2]; unsigned short u[16]; };

__device__ __forceinline__ unsigned short f2bf(float f) {
    unsigned int u = __float_as_uint(f);
    unsigned int r = (u + 0x7FFFu + ((u >> 16) & 1u)) >> 16;   // RNE
    return (unsigned short)r;
}
__device__ __forceinline__ float sigf(float x) { return 1.0f / (1.0f + __expf(-x)); }

// A-tile: 16 contiguous bf16 per lane (k = 16*half + s)
__device__ __forceinline__ V16 ldA16i(const unsigned short* __restrict__ base, unsigned off) {
    V16 x;
    x.q[0] = *(const v4u*)(base + off);
    x.q[1] = *(const v4u*)(base + off + 8);
    return x;
}
// B-tile: 16 contiguous bf16 per lane (K-major weight row)
__device__ __forceinline__ V16 ldB16i(const unsigned short* __restrict__ base, unsigned off) {
    V16 x;
    x.q[0] = *(const v4u*)(base + off);
    x.q[1] = *(const v4u*)(base + off + 8);
    return x;
}
__device__ __forceinline__ v8f wmma_bf16(const V16& A, const V16& B, v8f C) {
    return __builtin_amdgcn_wmma_f32_16x16x32_bf16(false, A.b, false, B.b,
                                                   (short)0, C, false, false);
}

// ---------------- constants ----------------
#define CB   32      // batch
#define CK   5       // beam width
#define CT   256     // encoder timesteps
#define CH   1024    // hidden
#define CH2  2048    // 2*hidden
#define CG4  4096    // 4*hidden
#define CV   10000   // vocab
#define CVP  10240   // vocab padded to tile multiple
#define CML  55      // max length
#define CNR  160     // B*K rows

// ---------------- one-time conversions ----------------
__global__ void f32_to_bf16_kernel(const float* __restrict__ s,
                                   unsigned short* __restrict__ d, long long n) {
    long long i = (long long)blockIdx.x * blockDim.x + threadIdx.x;
    long long stride = (long long)gridDim.x * blockDim.x;
    for (; i < n; i += stride) d[i] = f2bf(s[i]);
}

// W_out (2H x V) fp32 -> Wt (VP x 2H) bf16, zero-padded rows
__global__ void wout_transpose_kernel(const float* __restrict__ W,
                                      unsigned short* __restrict__ Wt) {
    long long n = (long long)CVP * CH2;
    long long i = (long long)blockIdx.x * blockDim.x + threadIdx.x;
    long long stride = (long long)gridDim.x * blockDim.x;
    for (; i < n; i += stride) {
        int vrow = (int)(i / CH2);
        int kcol = (int)(i % CH2);
        Wt[i] = (vrow < CV) ? f2bf(W[(long long)kcol * CV + vrow]) : (unsigned short)0;
    }
}

// enc (B,T,H) fp32 -> encT (B,H,T) bf16  (K-contiguous B-operand for ctx GEMM)
__global__ void enc_transpose_kernel(const float* __restrict__ enc,
                                     unsigned short* __restrict__ encT) {
    long long n = (long long)CB * CT * CH;
    long long i = (long long)blockIdx.x * blockDim.x + threadIdx.x;
    long long stride = (long long)gridDim.x * blockDim.x;
    for (; i < n; i += stride) {
        int b = (int)(i / ((long long)CT * CH));
        int rem = (int)(i % ((long long)CT * CH));
        int t = rem / CH, h = rem % CH;
        encT[((long long)b * CH + h) * CT + t] = f2bf(enc[i]);
    }
}

__global__ void pad_bias_kernel(const float* __restrict__ b, float* __restrict__ bp) {
    int i = blockIdx.x * blockDim.x + threadIdx.x;
    if (i < CVP) bp[i] = (i < CV) ? b[i] : -1.0e30f;
}

// zero h (bf16, in hc buffer) and c, set tokens = SOS(1) for first 32 rows
__global__ void init_state_kernel(unsigned short* __restrict__ hc,
                                  float* __restrict__ c, int* __restrict__ tokens) {
    int row = blockIdx.x;
    for (int e = threadIdx.x; e < CH; e += blockDim.x) {
        hc[(size_t)row * CH2 + e] = 0;
        c[(size_t)row * CH + e]   = 0.0f;
    }
    if (threadIdx.x == 0) tokens[row] = 1;
}

// replicate rows 0..31 -> 32..159 (reference quirk: h_k[n] = h[n % 32])
__global__ void tile_state_kernel(unsigned short* __restrict__ hc, float* __restrict__ c) {
    int n = 32 + blockIdx.x;
    int s = n & 31;
    for (int e = threadIdx.x; e < CH; e += blockDim.x) {
        hc[(size_t)n * CH2 + e] = hc[(size_t)s * CH2 + e];
        c[(size_t)n * CH + e]   = c[(size_t)s * CH + e];
    }
}

// ---------------- per-step kernels ----------------
__global__ void embed_kernel(const int* __restrict__ tokens, const float* __restrict__ emb,
                             unsigned short* __restrict__ xb) {
    int row = blockIdx.x;
    int t = tokens[row];
    for (int e = threadIdx.x; e < CH; e += blockDim.x)
        xb[(size_t)row * CH + e] = f2bf(emb[(size_t)t * CH + e]);
}

// gates = x @ W_ih^T + h @ W_hh^T + b ; K=1024, 2 N-tiles/wave, dual A, 4 wmma/chunk
// grid = (CG4/16/2/8 = 16, ceil(M/16)), block 256
__global__ void wmma_gates_kernel(const unsigned short* __restrict__ A0,  // x_bf  ld CH
                                  const unsigned short* __restrict__ A1,  // hc_bf ld CH2
                                  const unsigned short* __restrict__ W0,  // wih (4096 x 1024)
                                  const unsigned short* __restrict__ W1,  // whh
                                  const float* __restrict__ bias,
                                  float* __restrict__ C, int Mrows) {
    const int lane = threadIdx.x & 31, wave = threadIdx.x >> 5;
    const int nt = (blockIdx.x * 8 + wave) * 2;
    const int mtile = blockIdx.y;
    const int col = lane & 15, half = lane >> 4;
    const int kb = half * 16, koff = half * 16;
    int rowA = mtile * 16 + col; if (rowA >= Mrows) rowA = Mrows - 1;  // clamp: rows >= M unstored

    const unsigned aoff0 = (unsigned)rowA * CH  + kb;
    const unsigned aoff1 = (unsigned)rowA * CH2 + kb;
    const unsigned woffa = (unsigned)(nt * 16 + col) * CH + koff;
    const unsigned woffb = woffa + 16u * CH;

    v8f acc0, acc1;
#pragma unroll
    for (int r = 0; r < 8; ++r) { acc0[r] = 0.0f; acc1[r] = 0.0f; }

#pragma unroll 2
    for (int kc = 0; kc < (CH >> 5); ++kc) {
        const unsigned o = (unsigned)kc * 32;
        V16 a0  = ldA16i(A0, aoff0 + o);
        V16 a1  = ldA16i(A1, aoff1 + o);
        V16 b00 = ldB16i(W0, woffa + o);
        V16 b01 = ldB16i(W0, woffb + o);
        V16 b10 = ldB16i(W1, woffa + o);
        V16 b11 = ldB16i(W1, woffb + o);
        acc0 = wmma_bf16(a0, b00, acc0);
        acc1 = wmma_bf16(a0, b01, acc1);
        acc0 = wmma_bf16(a1, b10, acc0);
        acc1 = wmma_bf16(a1, b11, acc1);
    }

    const int n0 = nt * 16 + col;
    float bv0 = bias[n0], bv1 = bias[n0 + 16];
#pragma unroll
    for (int r = 0; r < 8; ++r) {
        int rowc = mtile * 16 + half * 8 + r;
        if (rowc < Mrows) {
            C[(size_t)rowc * CG4 + n0]      = acc0[r] + bv0;
            C[(size_t)rowc * CG4 + n0 + 16] = acc1[r] + bv1;
        }
    }
}

// C[M,Npad] = A[M,K] @ W[Npad,K]^T + bias ; generic, 2 N-tiles/wave
// grid = (Npad/16/2/8, ceil(M/16)), block 256
__global__ void wmma_gemm1_kernel(const unsigned short* __restrict__ A, int lda,
                                  const unsigned short* __restrict__ W,
                                  const float* __restrict__ bias,
                                  float* __restrict__ C, int ldc, int Mrows, int K) {
    const int lane = threadIdx.x & 31, wave = threadIdx.x >> 5;
    const int nt = (blockIdx.x * 8 + wave) * 2;
    const int mtile = blockIdx.y;
    const int col = lane & 15, half = lane >> 4;
    const int kb = half * 16, koff = half * 16;
    int rowA = mtile * 16 + col; if (rowA >= Mrows) rowA = Mrows - 1;

    const unsigned aoff  = (unsigned)rowA * lda + kb;
    const unsigned woff0 = (unsigned)(nt * 16 + col) * K + koff;
    const unsigned woff1 = woff0 + 16u * K;

    v8f acc0, acc1;
#pragma unroll
    for (int r = 0; r < 8; ++r) { acc0[r] = 0.0f; acc1[r] = 0.0f; }

    const int kchunks = K >> 5;
#pragma unroll 4
    for (int kc = 0; kc < kchunks; ++kc) {
        const unsigned o = (unsigned)kc * 32;
        V16 a  = ldA16i(A, aoff + o);
        V16 b0 = ldB16i(W, woff0 + o);
        V16 b1 = ldB16i(W, woff1 + o);
        acc0 = wmma_bf16(a, b0, acc0);
        acc1 = wmma_bf16(a, b1, acc1);
    }

    const int n0 = nt * 16 + col;
    float bv0 = bias ? bias[n0] : 0.0f;
    float bv1 = bias ? bias[n0 + 16] : 0.0f;
#pragma unroll
    for (int r = 0; r < 8; ++r) {
        int rowc = mtile * 16 + half * 8 + r;
        if (rowc < Mrows) {
            C[(size_t)rowc * ldc + n0]      = acc0[r] + bv0;
            C[(size_t)rowc * ldc + n0 + 16] = acc1[r] + bv1;
        }
    }
}

// gates(row,4H) -> c,h ; h written bf16 into hc[:, 0:H]
__global__ void lstm_kernel(const float* __restrict__ gates, float* __restrict__ c,
                            unsigned short* __restrict__ hc) {
    int row = blockIdx.x;
    const float* g = gates + (size_t)row * CG4;
    for (int e = threadIdx.x; e < CH; e += blockDim.x) {
        float gi = g[e], gf = g[CH + e], gg = g[2 * CH + e], go = g[3 * CH + e];
        float cn = sigf(gf) * c[(size_t)row * CH + e] + sigf(gi) * tanhf(gg);
        c[(size_t)row * CH + e] = cn;
        hc[(size_t)row * CH2 + e] = f2bf(sigf(go) * tanhf(cn));
    }
}

// scores[n,t] = h[n] . enc[b,t,:] ; grid = (32, 1), 8 waves * 2 t-tiles = T/16
__global__ void wmma_scores_kernel(const unsigned short* __restrict__ hc,
                                   const unsigned short* __restrict__ encb,  // (B,T,H) bf16
                                   float* __restrict__ scores, int rpb) {
    const int lane = threadIdx.x & 31, wave = threadIdx.x >> 5;
    const int b = blockIdx.x;
    const int nt = wave * 2;
    const int col = lane & 15, half = lane >> 4;
    const int kb = half * 16, koff = half * 16;
    int m = col; if (m >= rpb) m = rpb - 1;   // clamp (garbage rows unstored)

    const unsigned aoff  = (unsigned)(b * rpb + m) * CH2 + kb;
    const unsigned woff0 = ((unsigned)b * CT + nt * 16 + col) * CH + koff;
    const unsigned woff1 = woff0 + 16u * CH;

    v8f acc0, acc1;
#pragma unroll
    for (int r = 0; r < 8; ++r) { acc0[r] = 0.0f; acc1[r] = 0.0f; }

#pragma unroll 4
    for (int kc = 0; kc < (CH >> 5); ++kc) {
        const unsigned o = (unsigned)kc * 32;
        V16 a  = ldA16i(hc, aoff + o);
        V16 b0 = ldB16i(encb, woff0 + o);
        V16 b1 = ldB16i(encb, woff1 + o);
        acc0 = wmma_bf16(a, b0, acc0);
        acc1 = wmma_bf16(a, b1, acc1);
    }

#pragma unroll
    for (int r = 0; r < 8; ++r) {
        int rowm = half * 8 + r;
        if (rowm < rpb) {
            scores[(size_t)(b * rpb + rowm) * CT + nt * 16 + col]      = acc0[r];
            scores[(size_t)(b * rpb + rowm) * CT + nt * 16 + col + 16] = acc1[r];
        }
    }
}

__global__ void softmax_kernel(const float* __restrict__ scores,
                               unsigned short* __restrict__ attnb) {
    __shared__ float red[256];
    int row = blockIdx.x, tid = threadIdx.x;
    float x = scores[(size_t)row * CT + tid];
    red[tid] = x; __syncthreads();
    for (int s = 128; s > 0; s >>= 1) { if (tid < s) red[tid] = fmaxf(red[tid], red[tid + s]); __syncthreads(); }
    float mx = red[0]; __syncthreads();
    float e = __expf(x - mx);
    red[tid] = e; __syncthreads();
    for (int s = 128; s > 0; s >>= 1) { if (tid < s) red[tid] += red[tid + s]; __syncthreads(); }
    attnb[(size_t)row * CT + tid] = f2bf(e / red[0]);
}

// ctx = attn @ enc[b] via encT (B,H,T); result bf16 into hc[:, H:2H]
// grid = (32, 4), 8 waves * 2 h-tiles; K = T = 256
__global__ void wmma_ctx_kernel(const unsigned short* __restrict__ attnb,
                                const unsigned short* __restrict__ encT,
                                unsigned short* __restrict__ hc, int rpb) {
    const int lane = threadIdx.x & 31, wave = threadIdx.x >> 5;
    const int b = blockIdx.x;
    const int nt = (blockIdx.y * 8 + wave) * 2;
    const int col = lane & 15, half = lane >> 4;
    const int kb = half * 16, koff = half * 16;
    int m = col; if (m >= rpb) m = rpb - 1;

    const unsigned aoff  = (unsigned)(b * rpb + m) * CT + kb;
    const unsigned woff0 = ((unsigned)b * CH + nt * 16 + col) * CT + koff;
    const unsigned woff1 = woff0 + 16u * CT;

    v8f acc0, acc1;
#pragma unroll
    for (int r = 0; r < 8; ++r) { acc0[r] = 0.0f; acc1[r] = 0.0f; }

#pragma unroll
    for (int kc = 0; kc < (CT >> 5); ++kc) {
        const unsigned o = (unsigned)kc * 32;
        V16 a  = ldA16i(attnb, aoff + o);
        V16 b0 = ldB16i(encT, woff0 + o);
        V16 b1 = ldB16i(encT, woff1 + o);
        acc0 = wmma_bf16(a, b0, acc0);
        acc1 = wmma_bf16(a, b1, acc1);
    }

#pragma unroll
    for (int r = 0; r < 8; ++r) {
        int rowm = half * 8 + r;
        if (rowm < rpb) {
            hc[(size_t)(b * rpb + rowm) * CH2 + CH + nt * 16 + col]      = f2bf(acc0[r]);
            hc[(size_t)(b * rpb + rowm) * CH2 + CH + nt * 16 + col + 16] = f2bf(acc1[r]);
        }
    }
}

// fused: per-row logsumexp + top-5 of logits; outputs log-prob adjusted top-5
__global__ void topk_lse_kernel(const float* __restrict__ logits,
                                float* __restrict__ ov, int* __restrict__ oi) {
    __shared__ float red[256];
    __shared__ float sv[256 * 5];
    __shared__ int   si[256 * 5];
    int row = blockIdx.x, tid = threadIdx.x;

    // pass 1: local max + local top-5
    float mx = -3.0e38f;
    float v[5]; int ix[5];
#pragma unroll
    for (int j = 0; j < 5; ++j) { v[j] = -3.0e38f; ix[j] = 0; }
    for (int e = tid; e < CVP; e += 256) {
        float x = logits[(size_t)row * CVP + e];
        mx = fmaxf(mx, x);
        if (x > v[4]) {
            int j = 4;
            while (j > 0 && x > v[j - 1]) { v[j] = v[j - 1]; ix[j] = ix[j - 1]; --j; }
            v[j] = x; ix[j] = e;
        }
    }
    red[tid] = mx; __syncthreads();
    for (int s = 128; s > 0; s >>= 1) { if (tid < s) red[tid] = fmaxf(red[tid], red[tid + s]); __syncthreads(); }
    mx = red[0]; __syncthreads();

    // pass 2: sum of exp
    float sum = 0.0f;
    for (int e = tid; e < CVP; e += 256) sum += __expf(logits[(size_t)row * CVP + e] - mx);
    red[tid] = sum; __syncthreads();
    for (int s = 128; s > 0; s >>= 1) { if (tid < s) red[tid] += red[tid + s]; __syncthreads(); }
    float rowoff = mx + __logf(red[0]);

    // tree-merge the per-thread top-5 lists
#pragma unroll
    for (int j = 0; j < 5; ++j) { sv[tid * 5 + j] = v[j]; si[tid * 5 + j] = ix[j]; }
    for (int s = 128; s > 0; s >>= 1) {
        __syncthreads();
        if (tid < s) {
            float mv[5]; int mi[5]; int pa = 0, pb = 0;
#pragma unroll
            for (int j = 0; j < 5; ++j) {
                float va = (pa < 5) ? sv[tid * 5 + pa] : -3.4e38f;
                float vb = (pb < 5) ? sv[(tid + s) * 5 + pb] : -3.4e38f;
                if (va >= vb) { mv[j] = va; mi[j] = (pa < 5) ? si[tid * 5 + pa] : 0; ++pa; }
                else          { mv[j] = vb; mi[j] = (pb < 5) ? si[(tid + s) * 5 + pb] : 0; ++pb; }
            }
#pragma unroll
            for (int j = 0; j < 5; ++j) { sv[tid * 5 + j] = mv[j]; si[tid * 5 + j] = mi[j]; }
        }
    }
    __syncthreads();
    if (tid < 5) {
        ov[(size_t)row * 5 + tid] = sv[tid] - rowoff;
        oi[(size_t)row * 5 + tid] = si[tid];
    }
}

// step0: cum_ps / tokens / beams[:, :, 0] from the 32-row topk
__global__ void init_beams_kernel(const float* __restrict__ tkv, const int* __restrict__ tki,
                                  float* __restrict__ cum, int* __restrict__ tokens,
                                  int* __restrict__ beams) {
    int b = blockIdx.x, tid = threadIdx.x;
    if (tid < 5) {
        cum[b * 5 + tid]    = tkv[(size_t)b * 5 + tid];
        tokens[b * 5 + tid] = tki[(size_t)b * 5 + tid];
    }
    __syncthreads();
    for (int e = tid; e < 5 * CML; e += blockDim.x) {
        int i = e / CML, s = e % CML;
        beams[(size_t)(b * 5 + i) * CML + s] = (s == 0) ? tki[(size_t)b * 5 + i] : 0;
    }
}

// merge k*k candidates, reorder beams, append token at slot di
__global__ void beam_combine_kernel(const float* __restrict__ tkv, const int* __restrict__ tki,
                                    float* __restrict__ cum, int* __restrict__ tokens,
                                    const int* __restrict__ bsrc, int* __restrict__ bdst, int di) {
    __shared__ float sp[5]; __shared__ int sprev[5]; __shared__ int stok[5];
    int b = blockIdx.x, tid = threadIdx.x;
    if (tid == 0) {
        float cp[5];
        for (int j = 0; j < 5; ++j) cp[j] = cum[b * 5 + j];
        float cand[25]; int used[25];
        for (int j = 0; j < 5; ++j)
            for (int jj = 0; jj < 5; ++jj)
                cand[j * 5 + jj] = tkv[(size_t)(b * 5 + j) * 5 + jj] + cp[j];
        for (int i = 0; i < 25; ++i) used[i] = 0;
        for (int sel = 0; sel < 5; ++sel) {
            int best = 0; float bv = -3.4e38f;
            for (int i = 0; i < 25; ++i)
                if (!used[i] && cand[i] > bv) { bv = cand[i]; best = i; }
            used[best] = 1;
            sp[sel] = bv; sprev[sel] = best / 5;
            stok[sel] = tki[(size_t)(b * 5 + best / 5) * 5 + (best % 5)];
        }
        for (int i = 0; i < 5; ++i) { cum[b * 5 + i] = sp[i]; tokens[b * 5 + i] = stok[i]; }
    }
    __syncthreads();
    for (int e = tid; e < 5 * CML; e += blockDim.x) {
        int i = e / CML, s = e % CML;
        int val = (s == di) ? stok[i] : bsrc[(size_t)(b * 5 + sprev[i]) * CML + s];
        bdst[(size_t)(b * 5 + i) * CML + s] = val;
    }
}

__global__ void write_out_kernel(const int* __restrict__ beams, const float* __restrict__ cum,
                                 float* __restrict__ out) {
    int e = blockIdx.x * blockDim.x + threadIdx.x;
    if (e < CNR * CML) out[e] = (float)beams[e];
    else if (e < CNR * CML + CNR) out[e] = cum[e - CNR * CML];
}

// ---------------------------------------------------------------------------
extern "C" void kernel_launch(void* const* d_in, const int* in_sizes, int n_in,
                              void* d_out, int out_size, void* d_ws, size_t ws_size,
                              hipStream_t stream) {
    (void)in_sizes; (void)n_in; (void)out_size; (void)ws_size;
    const float* enc    = (const float*)d_in[1];
    const float* emb    = (const float*)d_in[3];
    const float* W_ih   = (const float*)d_in[4];
    const float* W_hh   = (const float*)d_in[5];
    const float* b_lstm = (const float*)d_in[6];
    const float* W_out  = (const float*)d_in[7];
    const float* b_out  = (const float*)d_in[8];
    float* out = (float*)d_out;

    char* ws = (char*)d_ws;
    size_t off = 0;
    auto alloc = [&](size_t bytes) -> char* {
        char* p = ws + off; off = (off + bytes + 255) & ~(size_t)255; return p;
    };
    unsigned short* wih_bf  = (unsigned short*)alloc((size_t)CG4 * CH * 2);
    unsigned short* whh_bf  = (unsigned short*)alloc((size_t)CG4 * CH * 2);
    unsigned short* wout_t  = (unsigned short*)alloc((size_t)CVP * CH2 * 2);
    unsigned short* enc_bf  = (unsigned short*)alloc((size_t)CB * CT * CH * 2);
    unsigned short* enc_t   = (unsigned short*)alloc((size_t)CB * CT * CH * 2);
    float*          bias_pad= (float*)alloc((size_t)CVP * 4);
    unsigned short* x_bf    = (unsigned short*)alloc((size_t)CNR * CH * 2);
    unsigned short* hc_bf   = (unsigned short*)alloc((size_t)CNR * CH2 * 2);
    float*          c_st    = (float*)alloc((size_t)CNR * CH * 4);
    float*          gates   = (float*)alloc((size_t)CNR * CG4 * 4);
    float*          scoresb = (float*)alloc((size_t)CNR * CT * 4);
    unsigned short* attn_bf = (unsigned short*)alloc((size_t)CNR * CT * 2);
    float*          logits  = (float*)alloc((size_t)CNR * CVP * 4);
    float*          tkv     = (float*)alloc((size_t)CNR * 5 * 4);
    int*            tki     = (int*)alloc((size_t)CNR * 5 * 4);
    float*          cum     = (float*)alloc((size_t)CNR * 4);
    int*            tokens  = (int*)alloc((size_t)CNR * 4);
    int*            beamsA  = (int*)alloc((size_t)CNR * CML * 4);
    int*            beamsB  = (int*)alloc((size_t)CNR * CML * 4);

    // one-time weight / encoder conversion to bf16 (L2-resident afterwards)
    f32_to_bf16_kernel<<<2048, 256, 0, stream>>>(W_ih, wih_bf, (long long)CG4 * CH);
    f32_to_bf16_kernel<<<2048, 256, 0, stream>>>(W_hh, whh_bf, (long long)CG4 * CH);
    f32_to_bf16_kernel<<<4096, 256, 0, stream>>>(enc, enc_bf, (long long)CB * CT * CH);
    enc_transpose_kernel<<<4096, 256, 0, stream>>>(enc, enc_t);
    wout_transpose_kernel<<<4096, 256, 0, stream>>>(W_out, wout_t);
    pad_bias_kernel<<<(CVP + 255) / 256, 256, 0, stream>>>(b_out, bias_pad);
    init_state_kernel<<<CB, 256, 0, stream>>>(hc_bf, c_st, tokens);

    auto fwd = [&](int Mrows, int rpb) {
        int Mt = (Mrows + 15) / 16;
        embed_kernel<<<Mrows, 256, 0, stream>>>(tokens, emb, x_bf);
        wmma_gates_kernel<<<dim3(CG4 / 256, Mt), 256, 0, stream>>>(
            x_bf, hc_bf, wih_bf, whh_bf, b_lstm, gates, Mrows);
        lstm_kernel<<<Mrows, 256, 0, stream>>>(gates, c_st, hc_bf);
        wmma_scores_kernel<<<dim3(CB, 1), 256, 0, stream>>>(hc_bf, enc_bf, scoresb, rpb);
        softmax_kernel<<<Mrows, 256, 0, stream>>>(scoresb, attn_bf);
        wmma_ctx_kernel<<<dim3(CB, 4), 256, 0, stream>>>(attn_bf, enc_t, hc_bf, rpb);
        // logits = [h, ctx] @ W_out + b_out  (dominant GEMM, W_out^T bf16 in L2)
        wmma_gemm1_kernel<<<dim3(CVP / 256, Mt), 256, 0, stream>>>(
            hc_bf, CH2, wout_t, bias_pad, logits, CVP, Mrows, CH2);
        topk_lse_kernel<<<Mrows, 256, 0, stream>>>(logits, tkv, tki);
    };

    // step 0: 32 rows, one row per batch, h0=c0=0
    fwd(CB, 1);
    init_beams_kernel<<<CB, 64, 0, stream>>>(tkv, tki, cum, tokens, beamsA);
    tile_state_kernel<<<CNR - CB, 256, 0, stream>>>(hc_bf, c_st);

    // steps 1..54: 160 rows, 5 beams per batch
    for (int di = 1; di < CML; ++di) {
        fwd(CNR, CK);
        const int* bsrc = (di & 1) ? beamsA : beamsB;
        int*       bdst = (di & 1) ? beamsB : beamsA;
        beam_combine_kernel<<<CB, 64, 0, stream>>>(tkv, tki, cum, tokens, bsrc, bdst, di);
    }
    // di=54 (even) wrote beamsA
    write_out_kernel<<<(CNR * CML + CNR + 255) / 256, 256, 0, stream>>>(beamsA, cum, out);
}
